// TritonFlashCausalAttention_41961830482041
// MI455X (gfx1250) — matrix operand order repaired and checked by reading, not compiled, not run on hardware
//
#include <hip/hip_runtime.h>
#include <hip/hip_bf16.h>
#include <math.h>
#include <stdint.h>

// Problem constants (match reference)
#define BATCH   2
#define SEQ     2048
#define DMODEL  2048
#define NHEADS  16
#define HEADD   128
#define DQKV    (3 * DMODEL)       // 6144
#define MROWS   (BATCH * SEQ)      // 4096
#define BIAS_MAX 8.0f

typedef __attribute__((ext_vector_type(16))) __bf16 v16bf;
typedef __attribute__((ext_vector_type(8)))  __bf16 v8bf;
typedef __attribute__((ext_vector_type(8)))  float  v8f;
typedef __attribute__((ext_vector_type(4)))  int    v4i;

// Pointer-to-int4 in specific address spaces (builtin signature: vV4i*1V4i*3IiIi)
typedef v4i __attribute__((address_space(1)))* gv4i_p;   // global
typedef v4i __attribute__((address_space(3)))* lv4i_p;   // LDS

// --------------------------------------------------------------------------
// CDNA5 async global->LDS copy (16B per lane), guarded so compile never breaks
// --------------------------------------------------------------------------
#if defined(__has_builtin)
#if __has_builtin(__builtin_amdgcn_global_load_async_to_lds_b128)
#define HAVE_ASYNC_LDS 1
#endif
#endif

__device__ __forceinline__ void async_cp16(void* lds, const void* glob) {
#if defined(HAVE_ASYNC_LDS)
    __builtin_amdgcn_global_load_async_to_lds_b128(
        (gv4i_p)(uintptr_t)glob,   // global VA == generic VA
        (lv4i_p)(uintptr_t)lds,    // low 32 bits of generic LDS VA = LDS offset
        /*offset=*/0, /*cpol=*/0);
#else
    *(v8bf*)lds = *(const v8bf*)glob;
#endif
}

#if defined(HAVE_ASYNC_LDS)
#if __has_builtin(__builtin_amdgcn_s_wait_asynccnt)
#define WAIT_ASYNC(n) __builtin_amdgcn_s_wait_asynccnt(n)
#else
#define WAIT_ASYNC(n) asm volatile("s_wait_asynccnt %0" ::"i"(n) : "memory")
#endif
#else
#define WAIT_ASYNC(n) do {} while (0)
#endif

__device__ __forceinline__ v8f vzero8() {
    v8f z;
#pragma unroll
    for (int i = 0; i < 8; ++i) z[i] = 0.0f;
    return z;
}

// A-fragment (16x32 bf16, M x K). Per ISA: lanes 0-15 -> K {0..7,16..23},
// lanes 16-31 -> K {8..15,24..31}; row M = lane & 15.
__device__ __forceinline__ v16bf load_a_frag(const __bf16* tile, int ldm) {
    const int lane = threadIdx.x & 31;
    const int row  = lane & 15;
    const int kb   = (lane >> 4) * 8;
    const __bf16* p = tile + row * ldm + kb;
    v8bf lo = *(const v8bf*)(p);
    v8bf hi = *(const v8bf*)(p + 16);
    v16bf r;
#pragma unroll
    for (int i = 0; i < 8; ++i) { r[i] = lo[i]; r[i + 8] = hi[i]; }
    return r;
}

// B-fragment (32x16 bf16, K x N) sourced from Bt stored row-major as N x K.
// Lanes 0-15 hold K {0..7,16..23}, lanes 16-31 hold K {8..15,24..31}; column N = lane & 15.
__device__ __forceinline__ v16bf load_b_frag(const __bf16* tileT, int ldm) {
    const int lane = threadIdx.x & 31;
    const int col  = lane & 15;
    const int kb   = (lane >> 4) * 8;
    const __bf16* p = tileT + col * ldm + kb;
    v8bf lo = *(const v8bf*)(p);
    v8bf hi = *(const v8bf*)(p + 16);
    v16bf r;
#pragma unroll
    for (int i = 0; i < 8; ++i) { r[i] = lo[i]; r[i + 8] = hi[i]; }
    return r;
}

__device__ __forceinline__ v8f wmma_bf16(v16bf a, v16bf b, v8f c) {
    return __builtin_amdgcn_wmma_f32_16x16x32_bf16(
        /*neg_a=*/false, a, /*neg_b=*/false, b,
        /*c_mod=*/(short)0, c, /*reuse_a=*/false, /*reuse_b=*/false);
}

// ---------------------------------------------------------------------------
// f32 -> bf16 elementwise convert
// ---------------------------------------------------------------------------
__global__ __launch_bounds__(256) void cvt_f32_bf16(const float* __restrict__ in,
                                                    __bf16* __restrict__ out, int n) {
    for (int i = blockIdx.x * 256 + threadIdx.x; i < n; i += gridDim.x * 256)
        out[i] = (__bf16)in[i];
}

// ---------------------------------------------------------------------------
// NT bf16 GEMM: C[M,N] = A[M,K] * Bw[N,K]^T + bias[N]
// Block tile 128x128, 256 threads = 8 waves (2x4), K-tile = 32.
// LDS double-buffered; tiles staged with async global->LDS DMA so tile t+1
// streams in while WMMA consumes tile t.
// ---------------------------------------------------------------------------
template <bool OUT_BF16>
__global__ __launch_bounds__(256) void gemm_nt_bf16(const __bf16* __restrict__ A,
                                                    const __bf16* __restrict__ Bw,
                                                    const float* __restrict__ bias,
                                                    void* __restrict__ Cout,
                                                    int Mdim, int Ndim, int Kdim) {
    __shared__ __bf16 As[2][128][40];   // 128 x 32 (+8 pad), 16B-aligned rows
    __shared__ __bf16 Bs[2][128][40];

    const int tile_m = blockIdx.y * 128;
    const int tile_n = blockIdx.x * 128;
    const int tid  = threadIdx.x;
    const int lane = tid & 31;
    const int w    = tid >> 5;
    const int wm   = (w & 1) * 64;     // wave M offset within block tile
    const int wn   = (w >> 1) * 32;    // wave N offset within block tile

    v8f c[4][2];
#pragma unroll
    for (int i = 0; i < 4; ++i)
#pragma unroll
        for (int j = 0; j < 2; ++j) c[i][j] = vzero8();

    // Stage one K-slice (128x32 of A and of B) into LDS buffer `buf`.
    auto stage = [&](int buf, int k0) {
#pragma unroll
        for (int i = 0; i < 2; ++i) {
            const int idx = tid + 256 * i;
            const int row = idx >> 2;
            const int col = (idx & 3) * 8;
            async_cp16(&As[buf][row][col],
                       A + (size_t)(tile_m + row) * Kdim + k0 + col);
            async_cp16(&Bs[buf][row][col],
                       Bw + (size_t)(tile_n + row) * Kdim + k0 + col);
        }
    };

    const int nk = Kdim / 32;
    stage(0, 0);                                  // prologue: fill buffer 0

    for (int kt = 0; kt < nk; ++kt) {
        const int cur = kt & 1;
        if (kt + 1 < nk) {
            stage(cur ^ 1, 32 * (kt + 1));        // DMA next tile (overlaps WMMA)
            WAIT_ASYNC(4);                        // only the 4 just-issued remain
        } else {
            WAIT_ASYNC(0);
        }
        __syncthreads();                          // current tile visible to all

        v16bf af[4];
#pragma unroll
        for (int i = 0; i < 4; ++i)
            af[i] = load_a_frag(&As[cur][wm + 16 * i][0], 40);
        v16bf bfrag[2];
#pragma unroll
        for (int j = 0; j < 2; ++j)
            bfrag[j] = load_b_frag(&Bs[cur][wn + 16 * j][0], 40);
#pragma unroll
        for (int i = 0; i < 4; ++i)
#pragma unroll
            for (int j = 0; j < 2; ++j)
                c[i][j] = wmma_bf16(af[i], bfrag[j], c[i][j]);
        __syncthreads();                          // readers done before re-stage
    }

    // Epilogue: C layout -> row = r + 8*(lane>=16), col = lane & 15
    const int half = lane >> 4;
    const int coll = lane & 15;
#pragma unroll
    for (int i = 0; i < 4; ++i)
#pragma unroll
        for (int j = 0; j < 2; ++j) {
            const int gcol = tile_n + wn + 16 * j + coll;
            const float bv = bias[gcol];
#pragma unroll
            for (int r = 0; r < 8; ++r) {
                const int grow = tile_m + wm + 16 * i + r + 8 * half;
                const float v = c[i][j][r] + bv;
                if (OUT_BF16)
                    ((__bf16*)Cout)[(size_t)grow * Ndim + gcol] = (__bf16)v;
                else
                    ((float*)Cout)[(size_t)grow * Ndim + gcol] = v;
            }
        }
}

// ---------------------------------------------------------------------------
// Flash attention with ALiBi + causal mask.
// Grid: (SEQ/128, NHEADS, BATCH). 256 threads = 8 waves; wave w owns 16 Q rows.
// qkv layout per row: [Q(2048) | K(2048) | V(2048)], head h at h*128.
// Q and K tiles staged with async global->LDS DMA; V is transposed through
// registers into LDS (needed for contiguous B-fragments of P*V).
// ---------------------------------------------------------------------------
__global__ __launch_bounds__(256) void flash_attn_bf16(const __bf16* __restrict__ qkv,
                                                       __bf16* __restrict__ ctx) {
    __shared__ __bf16 Qs[128][136];     // 128 x 128 Q tile
    __shared__ __bf16 Ks[32][136];      // 32 kv rows x 128
    __shared__ __bf16 VTs[128][40];     // V^T: 128 hd x 32 kv
    __shared__ __bf16 Ps[8][16][40];    // per-wave P tile (16 x 32)

    const int qt = blockIdx.x, h = blockIdx.y, b = blockIdx.z;
    const int q0 = qt * 128;
    const int tid  = threadIdx.x;
    const int lane = tid & 31;
    const int w    = tid >> 5;
    const int half = lane >> 4;
    const int coll = lane & 15;

    // Stage Q tile via async DMA (2048 chunks of 8 halfs, 8 per thread)
#pragma unroll
    for (int i = 0; i < 8; ++i) {
        const int idx = tid + 256 * i;
        const int row = idx >> 4;
        const int col = (idx & 15) * 8;
        async_cp16(&Qs[row][col],
                   qkv + (size_t)(b * SEQ + q0 + row) * DQKV + h * HEADD + col);
    }

    const float scale = 0.08838834764831845f;            // 1/sqrt(128)
    const float slope = exp2f(-(BIAS_MAX / NHEADS) * (float)(h + 1));

    float mrow[8], lrow[8];
    v8f acc[8];
#pragma unroll
    for (int r = 0; r < 8; ++r) { mrow[r] = -1e30f; lrow[r] = 0.0f; }
#pragma unroll
    for (int n = 0; n < 8; ++n) acc[n] = vzero8();

    const int ntiles = q0 / 32 + 4;                      // causal bound
    for (int t = 0; t < ntiles; ++t) {
        const int kb = t * 32;
        __syncthreads();  // prior tile consumed before overwriting K/V tiles

        // Stage K tile via async DMA; stage V transposed through registers.
#pragma unroll
        for (int i = 0; i < 2; ++i) {
            const int idx = tid + 256 * i;
            const int row = idx >> 4;
            const int col = (idx & 15) * 8;
            async_cp16(&Ks[row][col],
                       qkv + (size_t)(b * SEQ + kb + row) * DQKV + DMODEL +
                           h * HEADD + col);
            v8bf vv = *(const v8bf*)(
                qkv + (size_t)(b * SEQ + kb + row) * DQKV + 2 * DMODEL +
                h * HEADD + col);
#pragma unroll
            for (int u = 0; u < 8; ++u) VTs[col + u][row] = vv[u];
        }
        WAIT_ASYNC(0);            // Q (first iter) + K landed in LDS
        __syncthreads();

        // S = Q * K^T  (16 x 32 per wave, as two 16x16 fragments)
        v8f s0 = vzero8(), s1 = vzero8();
#pragma unroll
        for (int d0 = 0; d0 < 4; ++d0) {
            v16bf aq = load_a_frag(&Qs[16 * w][d0 * 32], 136);
            s0 = wmma_bf16(aq, load_b_frag(&Ks[0][d0 * 32], 136), s0);
            s1 = wmma_bf16(aq, load_b_frag(&Ks[16][d0 * 32], 136), s1);
        }

        // Online softmax (per-row stats reduced across each 16-lane half)
        float p0[8], p1[8];
#pragma unroll
        for (int r = 0; r < 8; ++r) {
            const int qrow = q0 + 16 * w + r + 8 * half;
            const int c0 = kb + coll;
            const int c1 = kb + 16 + coll;
            float sv0 = s0[r] * scale + (float)(c0 - (SEQ - 1)) * slope;
            float sv1 = s1[r] * scale + (float)(c1 - (SEQ - 1)) * slope;
            if (c0 > qrow) sv0 = -1e30f;
            if (c1 > qrow) sv1 = -1e30f;
            float tm = fmaxf(sv0, sv1);
#pragma unroll
            for (int off = 8; off >= 1; off >>= 1)
                tm = fmaxf(tm, __shfl_xor(tm, off, 32));
            const float mnew  = fmaxf(mrow[r], tm);
            const float alpha = __expf(mrow[r] - mnew);
            mrow[r] = mnew;
            const float e0 = __expf(sv0 - mnew);
            const float e1 = __expf(sv1 - mnew);
            float ts = e0 + e1;
#pragma unroll
            for (int off = 8; off >= 1; off >>= 1)
                ts += __shfl_xor(ts, off, 32);
            lrow[r] = lrow[r] * alpha + ts;
            p0[r] = e0; p1[r] = e1;
#pragma unroll
            for (int n = 0; n < 8; ++n) acc[n][r] *= alpha;
        }

        // Reshape P into an A-fragment through wave-private LDS.
#pragma unroll
        for (int r = 0; r < 8; ++r) {
            Ps[w][r + 8 * half][coll]      = (__bf16)p0[r];
            Ps[w][r + 8 * half][16 + coll] = (__bf16)p1[r];
        }
        __builtin_amdgcn_wave_barrier();                  // compile-time fence
        asm volatile("s_wait_dscnt 0x0" ::: "memory");    // DS in-order per wave

        v16bf ap = load_a_frag(&Ps[w][0][0], 40);
#pragma unroll
        for (int n = 0; n < 8; ++n)
            acc[n] = wmma_bf16(ap, load_b_frag(&VTs[16 * n][0], 40), acc[n]);
    }

    // Epilogue: ctx[b, q, h*128 + d] = acc / l  (bf16 for the out-proj GEMM)
#pragma unroll
    for (int n = 0; n < 8; ++n)
#pragma unroll
        for (int r = 0; r < 8; ++r) {
            const int qrow = q0 + 16 * w + r + 8 * half;
            const float v = acc[n][r] / lrow[r];
            ctx[(size_t)(b * SEQ + qrow) * DMODEL + h * HEADD + 16 * n + coll] = (__bf16)v;
        }
}

// ---------------------------------------------------------------------------
// Host-side launch
// ---------------------------------------------------------------------------
extern "C" void kernel_launch(void* const* d_in, const int* in_sizes, int n_in,
                              void* d_out, int out_size, void* d_ws, size_t ws_size,
                              hipStream_t stream) {
    (void)in_sizes; (void)n_in; (void)out_size; (void)ws_size;
    const float* x     = (const float*)d_in[0];
    const float* Wqkv  = (const float*)d_in[1];
    const float* b_qkv = (const float*)d_in[2];
    const float* Wout  = (const float*)d_in[3];
    const float* b_out = (const float*)d_in[4];
    float* out = (float*)d_out;

    // Workspace layout (bf16 copies + intermediates), ~112 MB total.
    __bf16* x_bf    = (__bf16*)d_ws;                       // [4096, 2048]
    __bf16* wqkv_bf = x_bf    + (size_t)MROWS * DMODEL;    // [6144, 2048]
    __bf16* wout_bf = wqkv_bf + (size_t)DQKV * DMODEL;     // [2048, 2048]
    __bf16* qkv_bf  = wout_bf + (size_t)DMODEL * DMODEL;   // [4096, 6144]
    __bf16* ctx_bf  = qkv_bf  + (size_t)MROWS * DQKV;      // [4096, 2048]

    cvt_f32_bf16<<<4096, 256, 0, stream>>>(x,    x_bf,    MROWS * DMODEL);
    cvt_f32_bf16<<<4096, 256, 0, stream>>>(Wqkv, wqkv_bf, DQKV * DMODEL);
    cvt_f32_bf16<<<4096, 256, 0, stream>>>(Wout, wout_bf, DMODEL * DMODEL);

    // QKV projection: [4096,6144] = x_bf [4096,2048] * Wqkv^T + b_qkv (bf16 out)
    gemm_nt_bf16<true><<<dim3(DQKV / 128, MROWS / 128), 256, 0, stream>>>(
        x_bf, wqkv_bf, b_qkv, (void*)qkv_bf, MROWS, DQKV, DMODEL);

    // Flash attention with ALiBi + causal mask
    flash_attn_bf16<<<dim3(SEQ / 128, NHEADS, BATCH), 256, 0, stream>>>(qkv_bf, ctx_bf);

    // Output projection: out = ctx * Wout^T + b_out (f32 out)
    gemm_nt_bf16<false><<<dim3(DMODEL / 128, MROWS / 128), 256, 0, stream>>>(
        ctx_bf, wout_bf, b_out, (void*)out, MROWS, DMODEL, DMODEL);
}
